// RNNBase_47132971107242
// MI455X (gfx1250) — matrix operand order repaired and checked by reading, not compiled, not run on hardware
//
#include <hip/hip_runtime.h>
#include <hip/hip_bf16.h>
#include <math.h>
#include <stdint.h>

// ---------------------------------------------------------------------------
// GRU-with-resets + Linear+ELU for MI455X (gfx1250, wave32, WMMA 16x16x32 f16)
// B=1024, T=256, D=128, H=128, G=3H=384
// ---------------------------------------------------------------------------

#define BB   1024
#define TT   256
#define DD   128
#define HH   128
#define GG   384   // 3*H
#define GRID_A 1024

typedef _Float16 v16h __attribute__((ext_vector_type(16)));
typedef _Float16 half8 __attribute__((ext_vector_type(8)));
typedef float    v8f  __attribute__((ext_vector_type(8)));

union H16 { v16h v; half8 h[2]; };

// D = A(16x32 f16) * B(32x16 f16) + C(16x16 f32)
static __device__ __forceinline__ v8f wmma_f16(v16h a, v16h b, v8f c) {
  return __builtin_amdgcn_wmma_f32_16x16x32_f16(
      /*neg_a=*/false, a, /*neg_b=*/false, b,
      /*c_mod=*/(short)0, c, /*reuse_a=*/false, /*reuse_b=*/false);
}

// CDNA5 async global->LDS copy (16B per lane), tracked by ASYNCcnt.
static __device__ __forceinline__ void async_copy_b128(uint32_t lds_off,
                                                       const void* gaddr) {
  asm volatile("global_load_async_to_lds_b128 %0, %1, off"
               :: "v"(lds_off), "v"(gaddr) : "memory");
}
static __device__ __forceinline__ void wait_asynccnt_le3() {
  asm volatile("s_wait_asynccnt 0x3" ::: "memory");
}

// Flat shared pointer -> 32-bit LDS byte offset (low 32 bits of the aperture).
static __device__ __forceinline__ uint32_t lds_addr(const void* p) {
  return (uint32_t)(uintptr_t)p;
}

// A-operand fragment from an LDS tile laid out as [16 rows][128 halves].
// ISA layout (16-bit A 16x32): lane%16 = M row; lane/16 selects K-halves:
//   elems 0..7  -> K = ks*32 + (lane/16)*8 + e
//   elems 8..15 -> K = ks*32 + 16 + (lane/16)*8 + (e-8)
static __device__ __forceinline__ v16h load_a_frag(const _Float16* hbase,
                                                   int lane, int ks) {
  int m  = lane & 15;
  int kh = (lane >> 4) & 1;
  const _Float16* p = hbase + m * DD + ks * 32 + kh * 8;
  H16 u;
  u.h[0] = *(const half8*)(p);
  u.h[1] = *(const half8*)(p + 16);
  return u.v;
}

// B-operand fragment gathered from global row-major f32 weights W[g][k].
// ISA layout (16-bit B 32x16): N = lane%16, K = ks*32 + (lane/16)*8 + (e&7) + (e>>3)*16.
static __device__ __forceinline__ v16h load_b_frag_gf32(const float* W, int ldk,
                                                        int g0, int ks, int lane) {
  int n  = lane & 15;
  int kh = (lane >> 4) & 1;
  const float* p = W + (size_t)(g0 + n) * ldk + ks * 32 + kh * 8;
  v16h b;
#pragma unroll
  for (int e = 0; e < 16; ++e) b[e] = (_Float16)p[(e & 7) + ((e >> 3) << 4)];
  return b;
}

// ---------------------------------------------------------------------------
// Kernel A: xi[t][b][g] = f16( x[b,t,:] . W_ih[g,:] + b_ih[g] )
// Block = 256 threads (8 waves). Wave w owns n-tiles {w, w+8, w+16}
// with W_ih fragments resident in VGPRs. Results are transposed through LDS
// so the 201 MB of xi goes out as coalesced b128 stores.
// ---------------------------------------------------------------------------
__global__ void __launch_bounds__(256)
gru_xproj_kernel(const float* __restrict__ x, const float* __restrict__ W_ih,
                 const float* __restrict__ b_ih, _Float16* __restrict__ xi) {
  __shared__ __align__(16) _Float16 stage[16 * DD];    // 16x128 x-tile as f16
  __shared__ __align__(16) _Float16 ostage[16 * GG];   // 16x384 result tile

  const int tid  = threadIdx.x;
  const int lane = tid & 31;
  const int wave = tid >> 5;

  // Preload W_ih fragments for this wave's 3 n-tiles (12 frags = 96 VGPRs).
  v16h Wf[3][4];
  float bias[3];
#pragma unroll
  for (int j = 0; j < 3; ++j) {
    int g0 = (wave + 8 * j) * 16;
#pragma unroll
    for (int ks = 0; ks < 4; ++ks)
      Wf[j][ks] = load_b_frag_gf32(W_ih, DD, g0, ks, lane);
    bias[j] = b_ih[g0 + (lane & 15)];
  }

  const int mrow = tid >> 4;    // 0..15 : staging row
  const int seg  = tid & 15;    // 8-float segment within the row
  const int mb   = (lane >> 4) * 8;

  for (int it = 0; it < 16; ++it) {
    int mt = it * GRID_A + blockIdx.x;           // 16384 M-tiles total
    // ---- stage one 16x128 tile of x as f16 (coalesced float4 loads) ----
    const float* xr = x + (size_t)(mt * 16 + mrow) * DD + seg * 8;
    float4 f0 = *(const float4*)(xr);
    float4 f1 = *(const float4*)(xr + 4);
    _Float16* sp = stage + mrow * DD + seg * 8;
    sp[0] = (_Float16)f0.x; sp[1] = (_Float16)f0.y;
    sp[2] = (_Float16)f0.z; sp[3] = (_Float16)f0.w;
    sp[4] = (_Float16)f1.x; sp[5] = (_Float16)f1.y;
    sp[6] = (_Float16)f1.z; sp[7] = (_Float16)f1.w;
    __syncthreads();                             // B1: x-tile staged

    v16h A[4];
#pragma unroll
    for (int ks = 0; ks < 4; ++ks) A[ks] = load_a_frag(stage, lane, ks);

#pragma unroll
    for (int j = 0; j < 3; ++j) {
      v8f acc;
#pragma unroll
      for (int i = 0; i < 8; ++i) acc[i] = bias[j];
#pragma unroll
      for (int ks = 0; ks < 4; ++ks) acc = wmma_f16(A[ks], Wf[j][ks], acc);

      int g = (wave + 8 * j) * 16 + (lane & 15);
#pragma unroll
      for (int r = 0; r < 8; ++r)
        ostage[(mb + r) * GG + g] = (_Float16)acc[r];
    }
    __syncthreads();                             // B2: result tile staged

    // ---- coalesced b128 stores: 768 segments of 8 halves, 3 per thread ----
#pragma unroll
    for (int k = 0; k < 3; ++k) {
      int s    = tid + k * 256;                  // 0..767
      int row  = s / 48;                         // 48 segments per row
      int off8 = (s % 48) * 8;
      half8 v  = *(const half8*)(ostage + row * GG + off8);
      int grow = mt * 16 + row;                  // flattened (b,t): row = b*T+t
      int b    = grow >> 8;                      // T = 256
      int t    = grow & 255;
      *(half8*)(xi + ((size_t)t * BB + b) * GG + off8) = v;
    }
    __syncthreads();                             // B3: safe to reuse staging
  }
}

// ---------------------------------------------------------------------------
// Kernel B: sequential scan over T with fused FC+ELU.
// 64 blocks, each owns a 16-batch tile; 8 waves split the 24 gate n-tiles
// (wave w -> columns w*16 of r, z, n) and 8 FC n-tiles (one each).
// W_hh / fc_W fragments live in VGPRs for the whole scan; the 12 KB xi slab
// for step t+1 is double-buffered into LDS with async b128 copies that
// overlap step t's WMMAs.
// ---------------------------------------------------------------------------
__global__ void __launch_bounds__(256)
gru_scan_kernel(const _Float16* __restrict__ xi, const float* __restrict__ h0,
                const int* __restrict__ done, const float* __restrict__ W_hh,
                const float* __restrict__ b_hh, const float* __restrict__ fcW,
                const float* __restrict__ fcb, float* __restrict__ out) {
  __shared__ __align__(16) float    hs32[16 * HH];       // h, f32 (gate math)
  __shared__ __align__(16) _Float16 hs16[16 * HH];       // h, f16 (WMMA A)
  __shared__ __align__(16) _Float16 xist[2][16 * GG];    // double-buffered xi_t

  const int tid  = threadIdx.x;
  const int lane = tid & 31;
  const int wave = tid >> 5;
  const int b0   = blockIdx.x * 16;

  // Async-copy geometry: 16*384 halves = 768 b128 segments -> 3 per thread.
  const int s0   = tid;            // segment ids: s0, s0+256, s0+512
  // Pre-split each segment id into (row, byte offset within row).
  int c_row[3], c_off[3];
#pragma unroll
  for (int k = 0; k < 3; ++k) {
    int s    = s0 + k * 256;
    c_row[k] = s / 48;
    c_off[k] = (s % 48) * 16;      // bytes within a 768B row
  }

  // ---- prologue: kick off async copy of xi[t=0] into buffer 0 ----
#pragma unroll
  for (int k = 0; k < 3; ++k) {
    const char* g = (const char*)(xi + ((size_t)0 * BB + b0 + c_row[k]) * GG)
                    + c_off[k];
    async_copy_b128(lds_addr((const char*)&xist[0][0] + c_row[k] * GG * 2 + c_off[k]), g);
  }

  // ---- init hidden state tile (overlaps the async copy) ----
  for (int i = tid; i < 16 * HH; i += 256) {
    int m = i >> 7, n = i & 127;
    float v = h0[(size_t)(b0 + m) * HH + n];
    hs32[i] = v;
    hs16[i] = (_Float16)v;
  }

  // ---- resident weight fragments: W_hh r/z/n tiles + fc_W tile ----
  const int g0 = wave * 16;
  v16h Wr[4], Wz[4], Wn[4], Wf[4];
#pragma unroll
  for (int ks = 0; ks < 4; ++ks) {
    Wr[ks] = load_b_frag_gf32(W_hh, HH, g0,          ks, lane);
    Wz[ks] = load_b_frag_gf32(W_hh, HH, g0 + HH,     ks, lane);
    Wn[ks] = load_b_frag_gf32(W_hh, HH, g0 + 2 * HH, ks, lane);
    Wf[ks] = load_b_frag_gf32(fcW,  HH, g0,          ks, lane);
  }
  const float bhr = b_hh[g0 + (lane & 15)];
  const float bhz = b_hh[HH + g0 + (lane & 15)];
  const float bhn = b_hh[2 * HH + g0 + (lane & 15)];
  const float bfc = fcb[g0 + (lane & 15)];

  const int nloc = g0 + (lane & 15);   // hidden-unit / gate column
  const int mb   = (lane >> 4) * 8;    // C-fragment row base
  const int mA   = lane & 15;          // A-fragment row (one row per lane; its k split over lanes m, m+16)

  __syncthreads();

  for (int t = 0; t < TT; ++t) {
    // ---- issue async copy of the next xi slab into the other buffer ----
    const int tn  = (t + 1 < TT) ? (t + 1) : (TT - 1);  // last issue is dummy
    const int nb  = (t + 1) & 1;
#pragma unroll
    for (int k = 0; k < 3; ++k) {
      const char* g = (const char*)(xi + ((size_t)tn * BB + b0 + c_row[k]) * GG)
                      + c_off[k];
      async_copy_b128(lds_addr((const char*)&xist[nb][0] + c_row[k] * GG * 2 + c_off[k]), g);
    }
    wait_asynccnt_le3();               // current slab (issued last step) done
    __syncthreads();                   // B1: slab visible to all waves

    const _Float16* xib = &xist[t & 1][0];

    // ---- reset factors (reset before step t uses done[:, t-1]) ----
    float rstA = 1.0f;
    if (t > 0) rstA = done[(size_t)(b0 + mA) * TT + (t - 1)] ? 0.0f : 1.0f;
    float rstH[8];
#pragma unroll
    for (int r = 0; r < 8; ++r)
      rstH[r] = (t > 0 && done[(size_t)(b0 + mb + r) * TT + (t - 1)]) ? 0.0f : 1.0f;

    // ---- A fragments of h (reset folded in: A-rows are lane-local) ----
    v16h A[4];
    const _Float16 sA = (_Float16)rstA;
#pragma unroll
    for (int ks = 0; ks < 4; ++ks)
      A[ks] = load_a_frag(hs16, lane, ks) * sA;     // v_pk_mul_f16 splat

    float hold[8];
#pragma unroll
    for (int r = 0; r < 8; ++r)
      hold[r] = hs32[(mb + r) * HH + nloc] * rstH[r];

    // ---- xi_t slice for this wave's columns (LDS f16 -> f32) ----
    float xr_[8], xz_[8], xn_[8];
#pragma unroll
    for (int r = 0; r < 8; ++r) {
      const _Float16* p = xib + (mb + r) * GG;
      xr_[r] = (float)p[nloc];
      xz_[r] = (float)p[HH + nloc];
      xn_[r] = (float)p[2 * HH + nloc];
    }

    // ---- gh = h @ W_hh^T + b_hh  (12 WMMAs) ----
    v8f ar, az, an;
#pragma unroll
    for (int i = 0; i < 8; ++i) { ar[i] = bhr; az[i] = bhz; an[i] = bhn; }
#pragma unroll
    for (int ks = 0; ks < 4; ++ks) {
      ar = wmma_f16(A[ks], Wr[ks], ar);
      az = wmma_f16(A[ks], Wz[ks], az);
      an = wmma_f16(A[ks], Wn[ks], an);
    }
    __syncthreads();   // B2: all reads of hs16/hs32 complete before updates

    // ---- gates + state update ----
    float hnew[8];
#pragma unroll
    for (int r = 0; r < 8; ++r) {
      float rr = 1.0f / (1.0f + expf(-(xr_[r] + ar[r])));
      float zz = 1.0f / (1.0f + expf(-(xz_[r] + az[r])));
      float nn = tanhf(xn_[r] + rr * an[r]);
      hnew[r]  = (1.0f - zz) * nn + zz * hold[r];
    }
#pragma unroll
    for (int r = 0; r < 8; ++r) {
      hs32[(mb + r) * HH + nloc] = hnew[r];
      hs16[(mb + r) * HH + nloc] = (_Float16)hnew[r];
    }
    __syncthreads();   // B3: updates visible before FC / next-step reads

    // ---- fused FC + ELU on h_t (4 WMMAs) ----
    v16h Af[4];
#pragma unroll
    for (int ks = 0; ks < 4; ++ks) Af[ks] = load_a_frag(hs16, lane, ks);
    v8f af;
#pragma unroll
    for (int i = 0; i < 8; ++i) af[i] = bfc;
#pragma unroll
    for (int ks = 0; ks < 4; ++ks) af = wmma_f16(Af[ks], Wf[ks], af);
#pragma unroll
    for (int r = 0; r < 8; ++r) {
      float v = af[r];
      float o = v > 0.0f ? v : (expf(v) - 1.0f);   // ELU(alpha=1)
      out[(size_t)(b0 + mb + r) * (TT * HH) + (size_t)t * HH + nloc] = o;
    }
  }

  __syncthreads();
  // ---- h_last -> second output region ----
  float* hlast = out + (size_t)BB * TT * HH;
  for (int i = tid; i < 16 * HH; i += 256) {
    int m = i >> 7, n = i & 127;
    hlast[(size_t)(b0 + m) * HH + n] = hs32[i];
  }
}

// ---------------------------------------------------------------------------
extern "C" void kernel_launch(void* const* d_in, const int* in_sizes, int n_in,
                              void* d_out, int out_size, void* d_ws, size_t ws_size,
                              hipStream_t stream) {
  const float* x      = (const float*)d_in[0];   // [B,T,D]
  const float* hidden = (const float*)d_in[1];   // [1,B,H]
  const int*   done   = (const int*)d_in[2];     // [B,T]
  const float* W_ih   = (const float*)d_in[3];   // [3H,D]
  const float* W_hh   = (const float*)d_in[4];   // [3H,H]
  const float* b_ih   = (const float*)d_in[5];   // [3H]
  const float* b_hh   = (const float*)d_in[6];   // [3H]
  const float* fcW    = (const float*)d_in[7];   // [H,H]
  const float* fcb    = (const float*)d_in[8];   // [H]
  float* out = (float*)d_out;                    // [B,T,H] ++ [1,B,H]

  _Float16* xi = (_Float16*)d_ws;                // [T,B,3H] f16 (201 MB)

  gru_xproj_kernel<<<GRID_A, 256, 0, stream>>>(x, W_ih, b_ih, xi);
  gru_scan_kernel<<<BB / 16, 256, 0, stream>>>(xi, hidden, done, W_hh, b_hh,
                                               fcW, fcb, out);
}
